// CloudGNN_72121090835129
// MI455X (gfx1250) — matrix operand order repaired and checked by reading, not compiled, not run on hardware
//
#include <hip/hip_runtime.h>
#include <hip/hip_bf16.h>

// ---------------------------------------------------------------------------
// CloudGNN for MI455X (gfx1250, wave32).
//   GCNConv -> GATConv(4 heads, mean) -> SAGEConv -> 2 MLP heads
// Dense GEMMs use v_wmma_f32_16x16x32_f16 (f32 accumulate); sparse segment
// ops use coalesced per-edge waves + f32 atomics (all accumulators live in
// the 192MB L2). Segment-max uses the monotonic float->uint order mapping
// with atomicMax (GLOBAL_ATOMIC_MAX_U32).
// ---------------------------------------------------------------------------

typedef __attribute__((ext_vector_type(16))) _Float16 v16h;
typedef __attribute__((ext_vector_type(8)))  float    v8f;

#define HDIM  64
#define HEADS 4

// ---- float -> uint monotonic order mapping (for atomic float max) ----------
__device__ __forceinline__ unsigned f2ord(float f) {
    unsigned u = __float_as_uint(f);
    return (u & 0x80000000u) ? ~u : (u | 0x80000000u);
}
__device__ __forceinline__ float ord2f(unsigned o) {
    unsigned u = (o & 0x80000000u) ? (o ^ 0x80000000u) : ~o;
    return __uint_as_float(u);
}
__device__ __forceinline__ float lrelu02(float v) { return v > 0.f ? v : 0.2f * v; }
__device__ __forceinline__ float sigm(float v) { return 1.f / (1.f + __expf(-v)); }

// ---- generic zero fill -----------------------------------------------------
__global__ void k_zero(unsigned* __restrict__ p, size_t n) {
    size_t i = (size_t)blockIdx.x * blockDim.x + threadIdx.x;
    size_t stride = (size_t)gridDim.x * blockDim.x;
    for (; i < n; i += stride) p[i] = 0u;
}

// ---------------------------------------------------------------------------
// WMMA GEMM: C[M x Nc] = A[M x K] * B[K x Nc] (+bias) (+relu) (+accumulate C)
// One wave -> one 16x16 tile. blockDim = 128 (4 waves); grid = (M/16, Nc/64).
// FLAGS (compile-time): bit0 = relu, bit1 = read existing C and accumulate.
// Requires K % 32 == 0, Nc % 64 == 0.
//
// Out-of-range rows are handled by CLAMPING load addresses (row m of A only
// feeds row m of D) so fragment loads stay unconditional and vectorize to
// global_load_b128. The epilogue takes a wave-UNIFORM full-tile fast path
// (scalar branch, no exec masking) -- with M % 16 == 0 every tile is full.
//
// Fragment layouts per CDNA5 ISA 7.12.2 (wave32):
//   A (16x32 f16): lane group g=lane>>4, m=lane&15; elem e<8 -> K=8g+e,
//                  e>=8 -> K=8+8g+e.
//   B (32x16 f16): col = lane&15; elem e<8 -> K = 8g + e, e>=8 -> K = 8+8g+e.
//   C (16x16 f32): col = lane&15; vgpr r -> row = r + 8g.
// ---------------------------------------------------------------------------
template <int FLAGS>
__global__ void __launch_bounds__(128)
k_wmma_gemm(const float* __restrict__ A, const float* __restrict__ B,
            const float* __restrict__ bias, float* __restrict__ C,
            int M, int K, int Nc) {
    const int wave = threadIdx.x >> 5;
    const int lane = threadIdx.x & 31;
    const int g = lane >> 4;
    const int t = lane & 15;
    const int mtile = blockIdx.x * 16;
    const int ntile = (blockIdx.y * 4 + wave) * 16;

    const int mrow   = mtile + t;
    const int mclamp = (mrow < M) ? mrow : (M - 1);
    const bool full  = (mtile + 16 <= M);          // wave-uniform

    float* const ctile = C + (size_t)(mtile + 8 * g) * Nc + (ntile + t);

    v8f acc = {};
    if (FLAGS & 2) {
        if (full) {
#pragma unroll
            for (int r = 0; r < 8; ++r) acc[r] = ctile[(size_t)r * Nc];
        } else {
#pragma unroll
            for (int r = 0; r < 8; ++r) {
                int m  = mtile + r + 8 * g;
                int mc = (m < M) ? m : (M - 1);
                acc[r] = C[(size_t)mc * Nc + ntile + t];
            }
        }
    }

    const float bv = bias ? bias[ntile + t] : 0.f;
    const float* arowbase = A + (size_t)mclamp * K + 8 * g;   // 32B aligned

    for (int k0 = 0; k0 < K; k0 += 32) {
        // A fragment: 4 unconditional float4 loads (-> global_load_b128)
        const float4* a4 = (const float4*)(arowbase + k0);
        float4 p0 = a4[0];   // K = k0+8g+0..3
        float4 p1 = a4[1];   // K = k0+8g+4..7
        float4 p2 = a4[4];   // K = k0+8g+16..19
        float4 p3 = a4[5];   // K = k0+8g+20..23
        v16h af;
        af[0]  = (_Float16)p0.x; af[1]  = (_Float16)p0.y;
        af[2]  = (_Float16)p0.z; af[3]  = (_Float16)p0.w;
        af[4]  = (_Float16)p1.x; af[5]  = (_Float16)p1.y;
        af[6]  = (_Float16)p1.z; af[7]  = (_Float16)p1.w;
        af[8]  = (_Float16)p2.x; af[9]  = (_Float16)p2.y;
        af[10] = (_Float16)p2.z; af[11] = (_Float16)p2.w;
        af[12] = (_Float16)p3.x; af[13] = (_Float16)p3.y;
        af[14] = (_Float16)p3.z; af[15] = (_Float16)p3.w;

        // B fragment: 16 column-strided loads (weights are tiny & L2-resident)
        const float* bcol = B + (size_t)(k0 + 8 * g) * Nc + ntile + t;
        v16h bf;
#pragma unroll
        for (int e = 0; e < 16; ++e)
            bf[e] = (_Float16)bcol[(size_t)(e < 8 ? e : e + 8) * Nc];

        acc = __builtin_amdgcn_wmma_f32_16x16x32_f16(
            /*neg_a=*/false, af, /*neg_b=*/false, bf,
            /*c_mod=*/(short)0, acc, /*reuse_a=*/false, /*reuse_b=*/false);
    }

    if (full) {
#pragma unroll
        for (int r = 0; r < 8; ++r) {
            float v = acc[r] + bv;
            if (FLAGS & 1) v = fmaxf(v, 0.f);
            ctile[(size_t)r * Nc] = v;
        }
    } else {
#pragma unroll
        for (int r = 0; r < 8; ++r) {
            int m = mtile + r + 8 * g;
            if (m < M) {
                float v = acc[r] + bv;
                if (FLAGS & 1) v = fmaxf(v, 0.f);
                C[(size_t)m * Nc + ntile + t] = v;
            }
        }
    }
}

// ---- degree (no self loops) ------------------------------------------------
__global__ void k_deg0(const long long* __restrict__ ei, float* __restrict__ deg0, int E) {
    int e = blockIdx.x * blockDim.x + threadIdx.x;
    if (e >= E) return;
    int c = (int)ei[(size_t)E + e];
    atomicAdd(&deg0[c], 1.f);
}

__global__ void k_dinv(const float* __restrict__ deg0, float* __restrict__ dinv, int N) {
    int n = blockIdx.x * blockDim.x + threadIdx.x;
    if (n >= N) return;
    dinv[n] = rsqrtf(deg0[n] + 1.f);   // self-loop augmented degree >= 1
}

// ---- GCN edge aggregation: acc[c] += dinv[r]*dinv[c]*hlin[r] ---------------
__global__ void k_gcn_edge(const long long* __restrict__ ei, const float* __restrict__ dinv,
                           const float* __restrict__ hlin, float* __restrict__ acc, int E) {
    int wid = (blockIdx.x * blockDim.x + threadIdx.x) >> 5;
    int lane = threadIdx.x & 31;
    if (wid >= E) return;
    int r = (int)ei[wid];
    int c = (int)ei[(size_t)E + wid];
    float norm = dinv[r] * dinv[c];
    const float* src = hlin + (size_t)r * HDIM;
    float* dst = acc + (size_t)c * HDIM;
    atomicAdd(&dst[lane], norm * src[lane]);
    atomicAdd(&dst[lane + 32], norm * src[lane + 32]);
}

// ---- GCN finish: h = relu(acc + dinv^2*hlin + b1) (in place on acc) --------
__global__ void k_gcn_finish(float* __restrict__ acc, const float* __restrict__ hlin,
                             const float* __restrict__ dinv, const float* __restrict__ b1, int N) {
    int t = blockIdx.x * blockDim.x + threadIdx.x;
    if (t >= N * HDIM) return;
    int n = t >> 6, f = t & 63;
    float di = dinv[n];
    float v = acc[t] + di * di * hlin[t] + b1[f];
    acc[t] = fmaxf(v, 0.f);
}

// ---- GAT per-node attention scores ----------------------------------------
__global__ void k_gat_scores(const float* __restrict__ hg, const float* __restrict__ att_src,
                             const float* __restrict__ att_dst, float* __restrict__ a_src,
                             float* __restrict__ a_dst, int N) {
    int t = blockIdx.x * blockDim.x + threadIdx.x;
    if (t >= N * HEADS) return;
    int n = t >> 2, h = t & 3;
    const float* row = hg + (size_t)n * (HEADS * HDIM) + h * HDIM;
    const float* as = att_src + h * HDIM;
    const float* ad = att_dst + h * HDIM;
    float s0 = 0.f, s1 = 0.f;
#pragma unroll 8
    for (int c = 0; c < HDIM; ++c) { s0 += row[c] * as[c]; s1 += row[c] * ad[c]; }
    a_src[t] = s0;
    a_dst[t] = s1;
}

// ---- GAT softmax max pass (edges + self loops) -----------------------------
__global__ void k_gat_max(const long long* __restrict__ ei, const float* __restrict__ a_src,
                          const float* __restrict__ a_dst, unsigned* __restrict__ emax,
                          int E, int N) {
    int idx = blockIdx.x * blockDim.x + threadIdx.x;
    if (idx >= (E + N) * HEADS) return;
    int eh = idx >> 2, h = idx & 3;
    int r, c;
    if (eh < E) { r = (int)ei[eh]; c = (int)ei[(size_t)E + eh]; } else { r = c = eh - E; }
    float ev = lrelu02(a_src[r * HEADS + h] + a_dst[c * HEADS + h]);
    atomicMax(&emax[c * HEADS + h], f2ord(ev));
}

// ---- GAT softmax denom pass ------------------------------------------------
__global__ void k_gat_denom(const long long* __restrict__ ei, const float* __restrict__ a_src,
                            const float* __restrict__ a_dst, const unsigned* __restrict__ emax,
                            float* __restrict__ denom, int E, int N) {
    int idx = blockIdx.x * blockDim.x + threadIdx.x;
    if (idx >= (E + N) * HEADS) return;
    int eh = idx >> 2, h = idx & 3;
    int r, c;
    if (eh < E) { r = (int)ei[eh]; c = (int)ei[(size_t)E + eh]; } else { r = c = eh - E; }
    float ev = lrelu02(a_src[r * HEADS + h] + a_dst[c * HEADS + h]);
    float ex = __expf(ev - ord2f(emax[c * HEADS + h]));
    atomicAdd(&denom[c * HEADS + h], ex);
}

// ---- GAT weighted aggregation: agg[c,h,:] += alpha * hg[r,h,:] -------------
__global__ void k_gat_agg(const long long* __restrict__ ei, const float* __restrict__ a_src,
                          const float* __restrict__ a_dst, const unsigned* __restrict__ emax,
                          const float* __restrict__ denom, const float* __restrict__ hg,
                          float* __restrict__ agg, int E, int N) {
    int wid = (blockIdx.x * blockDim.x + threadIdx.x) >> 5;
    int lane = threadIdx.x & 31;
    if (wid >= E + N) return;
    int r, c;
    if (wid < E) { r = (int)ei[wid]; c = (int)ei[(size_t)E + wid]; } else { r = c = wid - E; }
    float alpha[HEADS];
#pragma unroll
    for (int h = 0; h < HEADS; ++h) {
        float ev = lrelu02(a_src[r * HEADS + h] + a_dst[c * HEADS + h]);
        alpha[h] = __expf(ev - ord2f(emax[c * HEADS + h])) / denom[c * HEADS + h];
    }
    const float* src = hg + (size_t)r * (HEADS * HDIM);
    float* dst = agg + (size_t)c * (HEADS * HDIM);
#pragma unroll
    for (int h = 0; h < HEADS; ++h) {
        int f0 = h * HDIM + lane;
        atomicAdd(&dst[f0], alpha[h] * src[f0]);
        atomicAdd(&dst[f0 + 32], alpha[h] * src[f0 + 32]);
    }
}

// ---- GAT output: h2 = relu(mean_h(agg) + bg) -------------------------------
__global__ void k_gat_out(const float* __restrict__ agg, const float* __restrict__ bg,
                          float* __restrict__ h2, int N) {
    int t = blockIdx.x * blockDim.x + threadIdx.x;
    if (t >= N * HDIM) return;
    int n = t >> 6, f = t & 63;
    const float* a = agg + (size_t)n * (HEADS * HDIM);
    float s = 0.25f * (a[f] + a[HDIM + f] + a[2 * HDIM + f] + a[3 * HDIM + f]);
    h2[t] = fmaxf(s + bg[f], 0.f);
}

// ---- SAGE mean aggregation (no self loops) ---------------------------------
__global__ void k_sage_agg(const long long* __restrict__ ei, const float* __restrict__ h2,
                           float* __restrict__ macc, int E) {
    int wid = (blockIdx.x * blockDim.x + threadIdx.x) >> 5;
    int lane = threadIdx.x & 31;
    if (wid >= E) return;
    int r = (int)ei[wid];
    int c = (int)ei[(size_t)E + wid];
    const float* src = h2 + (size_t)r * HDIM;
    float* dst = macc + (size_t)c * HDIM;
    atomicAdd(&dst[lane], src[lane]);
    atomicAdd(&dst[lane + 32], src[lane + 32]);
}

__global__ void k_sage_norm(float* __restrict__ macc, const float* __restrict__ deg0, int N) {
    int t = blockIdx.x * blockDim.x + threadIdx.x;
    if (t >= N * HDIM) return;
    int n = t >> 6;
    macc[t] /= fmaxf(deg0[n], 1.f);
}

// ---- MLP heads: anomaly / risk from emb ------------------------------------
__global__ void k_heads(const float* __restrict__ emb,
                        const float* __restrict__ a1w, const float* __restrict__ a1b,
                        const float* __restrict__ a2w, const float* __restrict__ a2b,
                        const float* __restrict__ r1w, const float* __restrict__ r1b,
                        const float* __restrict__ r2w, const float* __restrict__ r2b,
                        float* __restrict__ anomaly, float* __restrict__ risk, int N) {
    int n = blockIdx.x * blockDim.x + threadIdx.x;
    if (n >= N) return;
    float er[HDIM];
#pragma unroll
    for (int k = 0; k < HDIM; ++k) er[k] = emb[(size_t)n * HDIM + k];

    float acc_a = a2b[0];
    for (int j = 0; j < 32; ++j) {
        float s = a1b[j];
#pragma unroll 8
        for (int k = 0; k < HDIM; ++k) s += er[k] * a1w[k * 32 + j];
        acc_a += fmaxf(s, 0.f) * a2w[j];
    }
    anomaly[n] = sigm(acc_a);

    float acc_r = r2b[0];
    for (int j = 0; j < 32; ++j) {
        float s = r1b[j];
#pragma unroll 8
        for (int k = 0; k < HDIM; ++k) s += er[k] * r1w[k * 32 + j];
        acc_r += fmaxf(s, 0.f) * r2w[j];
    }
    risk[n] = sigm(acc_r);
}

// ---------------------------------------------------------------------------
extern "C" void kernel_launch(void* const* d_in, const int* in_sizes, int n_in,
                              void* d_out, int out_size, void* d_ws, size_t ws_size,
                              hipStream_t stream) {
    const float*     x       = (const float*)d_in[0];
    const long long* ei      = (const long long*)d_in[1];
    const float*     W1      = (const float*)d_in[2];
    const float*     b1      = (const float*)d_in[3];
    const float*     Wg      = (const float*)d_in[4];
    const float*     att_src = (const float*)d_in[5];
    const float*     att_dst = (const float*)d_in[6];
    const float*     bg      = (const float*)d_in[7];
    const float*     Wl      = (const float*)d_in[8];
    const float*     bl      = (const float*)d_in[9];
    const float*     Wr      = (const float*)d_in[10];
    const float*     a1w     = (const float*)d_in[11];
    const float*     a1b     = (const float*)d_in[12];
    const float*     a2w     = (const float*)d_in[13];
    const float*     a2b     = (const float*)d_in[14];
    const float*     r1w     = (const float*)d_in[15];
    const float*     r1b     = (const float*)d_in[16];
    const float*     r2w     = (const float*)d_in[17];
    const float*     r2b     = (const float*)d_in[18];

    const int F_IN = 128;
    const int N = in_sizes[0] / F_IN;
    const int E = in_sizes[1] / 2;

    // ---- workspace layout (floats) ----
    float* ws = (float*)d_ws;
    size_t off = 0;
    float*    hlin  = ws + off; off += (size_t)N * HDIM;          // -> later mean_agg
    float*    hbuf  = ws + off; off += (size_t)N * HDIM;          // GCN acc/h -> GAT h2
    float*    hg    = ws + off; off += (size_t)N * HEADS * HDIM;
    float*    agg   = ws + off; off += (size_t)N * HEADS * HDIM;
    float*    a_src = ws + off; off += (size_t)N * HEADS;
    float*    a_dst = ws + off; off += (size_t)N * HEADS;
    unsigned* emax  = (unsigned*)(ws + off); off += (size_t)N * HEADS;
    float*    denom = ws + off; off += (size_t)N * HEADS;
    float*    deg0  = ws + off; off += (size_t)N;
    float*    dinv  = ws + off; off += (size_t)N;
    const size_t ws_floats = off;

    float* emb     = (float*)d_out;                 // N x 64
    float* anomaly = emb + (size_t)N * HDIM;        // N
    float* risk    = anomaly + N;                   // N

    const int TB = 256;
    auto cdiv = [](long long a, long long b) { return (int)((a + b - 1) / b); };

    // 0) zero all accumulators (emax==0 encodes -inf in order space)
    k_zero<<<4096, TB, 0, stream>>>((unsigned*)ws, ws_floats);

    // 1) hlin = x @ W1                       [N,128]x[128,64]
    k_wmma_gemm<0><<<dim3(cdiv(N, 16), 1), 128, 0, stream>>>(x, W1, nullptr, hlin, N, F_IN, HDIM);

    // 2) degrees & symmetric norm
    k_deg0<<<cdiv(E, TB), TB, 0, stream>>>(ei, deg0, E);
    k_dinv<<<cdiv(N, TB), TB, 0, stream>>>(deg0, dinv, N);

    // 3) GCN aggregation (one wave per edge) + finish (self-loop + bias + relu)
    k_gcn_edge<<<cdiv((long long)E * 32, TB), TB, 0, stream>>>(ei, dinv, hlin, hbuf, E);
    k_gcn_finish<<<cdiv((long long)N * HDIM, TB), TB, 0, stream>>>(hbuf, hlin, dinv, b1, N);

    // 4) hg = h @ Wg                         [N,64]x[64,256]
    k_wmma_gemm<0><<<dim3(cdiv(N, 16), 4), 128, 0, stream>>>(hbuf, Wg, nullptr, hg, N, HDIM, HEADS * HDIM);

    // 5) GAT attention: scores, segment-max, softmax denom, weighted agg
    k_gat_scores<<<cdiv((long long)N * HEADS, TB), TB, 0, stream>>>(hg, att_src, att_dst, a_src, a_dst, N);
    k_gat_max<<<cdiv((long long)(E + N) * HEADS, TB), TB, 0, stream>>>(ei, a_src, a_dst, emax, E, N);
    k_gat_denom<<<cdiv((long long)(E + N) * HEADS, TB), TB, 0, stream>>>(ei, a_src, a_dst, emax, denom, E, N);
    k_gat_agg<<<cdiv((long long)(E + N) * 32, TB), TB, 0, stream>>>(ei, a_src, a_dst, emax, denom, hg, agg, E, N);
    k_gat_out<<<cdiv((long long)N * HDIM, TB), TB, 0, stream>>>(agg, bg, hbuf, N);   // hbuf := h2

    // 6) SAGE: mean over in-neighbors (reuse hlin buffer as mean accumulator)
    k_zero<<<2048, TB, 0, stream>>>((unsigned*)hlin, (size_t)N * HDIM);
    k_sage_agg<<<cdiv((long long)E * 32, TB), TB, 0, stream>>>(ei, hbuf, hlin, E);
    k_sage_norm<<<cdiv((long long)N * HDIM, TB), TB, 0, stream>>>(hlin, deg0, N);

    // 7) emb = mean_agg @ Wl + bl + h2 @ Wr  (two WMMA passes, 2nd accumulates)
    k_wmma_gemm<0><<<dim3(cdiv(N, 16), 1), 128, 0, stream>>>(hlin, Wl, bl, emb, N, HDIM, HDIM);
    k_wmma_gemm<2><<<dim3(cdiv(N, 16), 1), 128, 0, stream>>>(hbuf, Wr, nullptr, emb, N, HDIM, HDIM);

    // 8) heads
    k_heads<<<cdiv(N, TB), TB, 0, stream>>>(emb, a1w, a1b, a2w, a2b, r1w, r1b, r2w, r2b,
                                            anomaly, risk, N);
}
